// RecurrentAttentionNetwork_2035814498871
// MI455X (gfx1250) — compile-verified, matrix operand order checked by reading, not run on hardware
//
#include <hip/hip_runtime.h>
#include <hip/hip_bf16.h>

// Problem dims (fixed by the reference)
#define BD 16
#define SD 2048
#define ID 1024
#define HD 1024
#define KT 32      // ID/32 = HD/32 k-tiles
#define NWG 16     // persistent grid: 16 workgroups
#define TPB 128    // 4 waves each -> 64 waves total == 64 16-col tiles of H (and of I)

typedef __attribute__((ext_vector_type(16))) __bf16 v16bf;
typedef __attribute__((ext_vector_type(8)))  float  v8f;

static __device__ __forceinline__ v8f wmma_bf16(v16bf a, v16bf b, v8f c) {
  // D = A(16x32 bf16) x B(32x16 bf16) + C(16x16 f32)
  return __builtin_amdgcn_wmma_f32_16x16x32_bf16(false, a, false, b, (short)0, c, false, false);
}
static __device__ __forceinline__ v16bf ldfrag(const __hip_bfloat16* p) {
  return *(const v16bf*)(const void*)p;
}
static __device__ __forceinline__ float sigm(float x) { return 1.0f / (1.0f + __expf(-x)); }
static __device__ __forceinline__ float tanh_fast(float x) { return 2.0f * sigm(2.0f * x) - 1.0f; }

// 16-bit A/B fragment map (ISA 7.12.2): lane L holds 16 bf16; laneHalf = L>>4.
// slot s (0..15)  ->  kk (0..31) inside the 32-wide k-tile.
static __device__ __forceinline__ int frag_kk(int s, int laneHalf) {
  return (s & 7) + (laneHalf << 3) + ((s >> 3) << 4);
}

// ---------------- prep: pack x[B,S,I] -> A-fragment tiles xp[t][kt][lane][16] ----------------
__global__ void pack_x_kernel(const float* __restrict__ x, __hip_bfloat16* __restrict__ xp) {
  int tile = (blockIdx.x * blockDim.x + threadIdx.x) >> 5;   // t*KT + kt
  int lane = threadIdx.x & 31;
  if (tile >= SD * KT) return;
  int t = tile / KT, kt = tile % KT;
  int m = lane & 15, hh = lane >> 4;                         // m = batch row
  __hip_bfloat16* dst = xp + (size_t)tile * 512 + lane * 16;
  const float* src = x + (size_t)m * SD * ID + (size_t)t * ID + kt * 32;
#pragma unroll
  for (int s = 0; s < 16; ++s)
    dst[s] = __float2bfloat16(src[frag_kk(s, hh)]);
}

// ---------------- prep: pack w[N,K] (row major) -> B-fragment tiles [nt][kt][lane][16] --------
__global__ void pack_w_kernel(const float* __restrict__ w, __hip_bfloat16* __restrict__ wp,
                              int N, int K) {
  int ktiles = K / 32;
  int tile = (blockIdx.x * blockDim.x + threadIdx.x) >> 5;   // nt*ktiles + kt
  int lane = threadIdx.x & 31;
  if (tile >= (N / 16) * ktiles) return;
  int nt = tile / ktiles, kt = tile % ktiles;
  int n = nt * 16 + (lane & 15), hh = lane >> 4;
  __hip_bfloat16* dst = wp + (size_t)tile * 512 + lane * 16;
  const float* src = w + (size_t)n * K + kt * 32;
#pragma unroll
  for (int s = 0; s < 16; ++s)
    dst[s] = __float2bfloat16(src[frag_kk(s, hh)]);
}

// ---------------- prep: zero barrier counter + h0 buffers, attn = softmax(singleton) = 1 ------
__global__ void init_kernel(unsigned* __restrict__ cnt, __hip_bfloat16* __restrict__ hA0,
                            float* __restrict__ hF0, float* __restrict__ attn) {
  int i = blockIdx.x * blockDim.x + threadIdx.x;
  if (i == 0) *cnt = 0u;
  if (i < BD * HD) { hF0[i] = 0.0f; hA0[i] = __float2bfloat16(0.0f); }
  if (i < BD * SD) attn[i] = 1.0f;
}

// monotonic device-wide barrier (grid of NWG is trivially resident)
static __device__ __forceinline__ void grid_barrier(unsigned* cnt, unsigned target) {
  __threadfence();
  __syncthreads();
  if (threadIdx.x == 0) {
    __hip_atomic_fetch_add(cnt, 1u, __ATOMIC_RELEASE, __HIP_MEMORY_SCOPE_AGENT);
    while (__hip_atomic_load(cnt, __ATOMIC_ACQUIRE, __HIP_MEMORY_SCOPE_AGENT) < target)
      __builtin_amdgcn_s_sleep(2);
  }
  __syncthreads();
}

// ---------------- persistent fused GRU scan + FC ---------------------------------------------
__global__ void __launch_bounds__(TPB)
gru_scan_kernel(const __hip_bfloat16* __restrict__ xp,
                const __hip_bfloat16* __restrict__ wihP,   // 192 n-tiles x 32 k-tiles
                const __hip_bfloat16* __restrict__ whhP,   // 192 x 32
                const __hip_bfloat16* __restrict__ fcP,    // 64 x 32
                const float* __restrict__ b_ih, const float* __restrict__ b_hh,
                const float* __restrict__ fc_b,
                __hip_bfloat16* __restrict__ hA0, __hip_bfloat16* __restrict__ hA1,
                float* __restrict__ hF0, float* __restrict__ hF1,
                float* __restrict__ out, float* __restrict__ hidden_out,
                unsigned* __restrict__ cnt) {
  const int lane = threadIdx.x & 31;
  const int wave = blockIdx.x * (TPB >> 5) + (threadIdx.x >> 5);   // 0..63: owns 16 cols
  const int col  = (wave << 4) + (lane & 15);                      // 0..1023
  const int lo   = lane * 16;

  // per-gate B-tile bases (n-tile index = gate*64 + wave)
  const __hip_bfloat16* wih_r = wihP + (size_t)((0   + wave) * KT) * 512;
  const __hip_bfloat16* wih_z = wihP + (size_t)((64  + wave) * KT) * 512;
  const __hip_bfloat16* wih_n = wihP + (size_t)((128 + wave) * KT) * 512;
  const __hip_bfloat16* whh_r = whhP + (size_t)((0   + wave) * KT) * 512;
  const __hip_bfloat16* whh_z = whhP + (size_t)((64  + wave) * KT) * 512;
  const __hip_bfloat16* whh_n = whhP + (size_t)((128 + wave) * KT) * 512;
  const __hip_bfloat16* fcw   = fcP  + (size_t)(wave * KT) * 512;

  const float br = b_ih[col]            + b_hh[col];
  const float bz = b_ih[HD + col]       + b_hh[HD + col];
  const float bni = b_ih[2 * HD + col];
  const float bnh = b_hh[2 * HD + col];
  const float bo  = fc_b[col];

  const int ktile_h = col >> 5;                      // where this wave's h cols live in A-frags
  const int kk      = col & 31;
  const int slot    = (kk & 7) + ((kk >> 4) << 3);
  const int plane   = (kk >> 3) & 1;                 // lane-half select for that kk

  for (int t = 0; t < SD; ++t) {
    const __hip_bfloat16* hAc = (t & 1) ? hA1 : hA0;
    __hip_bfloat16*       hAn = (t & 1) ? hA0 : hA1;
    const float*          hFc = (t & 1) ? hF1 : hF0;
    float*                hFn = (t & 1) ? hF0 : hF1;
    const __hip_bfloat16* xpt = xp + (size_t)t * KT * 512;

    v8f ar = {}, az = {}, ain = {}, ahn = {};
#pragma unroll 4
    for (int kt = 0; kt < KT; ++kt) {
      const int off = kt * 512 + lo;
      __builtin_prefetch(xpt + off + 512, 0, 0);     // next k-tile of packed x
      v16bf Ax = ldfrag(xpt + off);
      v16bf Ah = ldfrag(hAc + off);
      ar  = wmma_bf16(Ax, ldfrag(wih_r + off), ar);
      ar  = wmma_bf16(Ah, ldfrag(whh_r + off), ar);
      az  = wmma_bf16(Ax, ldfrag(wih_z + off), az);
      az  = wmma_bf16(Ah, ldfrag(whh_z + off), az);
      ain = wmma_bf16(Ax, ldfrag(wih_n + off), ain);
      ahn = wmma_bf16(Ah, ldfrag(whh_n + off), ahn);
    }

    // GRU elementwise: C/D layout -> VGPR v holds row m = v + 8*(lane>>4), col = lane&15 tile col
#pragma unroll
    for (int v = 0; v < 8; ++v) {
      const int m = v + ((lane >> 4) << 3);
      const float r  = sigm(ar[v] + br);
      const float z  = sigm(az[v] + bz);
      const float nn = tanh_fast(ain[v] + bni + r * (ahn[v] + bnh));
      const float hp = hFc[m * HD + col];
      const float hn = (1.0f - z) * nn + z * hp;
      hFn[m * HD + col] = hn;
      // scatter into next-step A-fragment layout
      const int fl = (m & 15) + (plane << 4);
      hAn[(size_t)ktile_h * 512 + fl * 16 + slot] = __float2bfloat16(hn);
      if (t == SD - 1) hidden_out[m * HD + col] = hn;
    }

    grid_barrier(cnt, (unsigned)(t + 1) * NWG);      // one device barrier per step

    // fused FC on the freshly produced h_t (reads hAn, written this step by all waves)
    v8f ao = {};
#pragma unroll 4
    for (int kt = 0; kt < KT; ++kt) {
      const int off = kt * 512 + lo;
      ao = wmma_bf16(ldfrag(hAn + off), ldfrag(fcw + off), ao);
    }
#pragma unroll
    for (int v = 0; v < 8; ++v) {
      const int m = v + ((lane >> 4) << 3);
      out[(size_t)m * SD * ID + (size_t)t * ID + col] = ao[v] + bo;
    }
  }
}

// ---------------------------------------------------------------------------------------------
extern "C" void kernel_launch(void* const* d_in, const int* in_sizes, int n_in,
                              void* d_out, int out_size, void* d_ws, size_t ws_size,
                              hipStream_t stream) {
  const float* x    = (const float*)d_in[0];
  // d_in[1..4] = attention weights: mathematically dead (softmax over a size-1 axis == 1)
  const float* w_ih = (const float*)d_in[5];
  const float* w_hh = (const float*)d_in[6];
  const float* b_ih = (const float*)d_in[7];
  const float* b_hh = (const float*)d_in[8];
  const float* fc_w = (const float*)d_in[9];
  const float* fc_b = (const float*)d_in[10];

  float* out    = (float*)d_out;                      // [B,S,I]
  float* hidden = out + (size_t)BD * SD * ID;         // [1,B,H]
  float* attn   = hidden + (size_t)BD * HD;           // [B,S,1]

  // workspace layout (all 256B aligned)
  char* ws = (char*)d_ws;
  unsigned*        cnt  = (unsigned*)ws;                         size_t o = 256;
  __hip_bfloat16*  xp   = (__hip_bfloat16*)(ws + o);             o += (size_t)SD * KT * 512 * 2;
  __hip_bfloat16*  wihP = (__hip_bfloat16*)(ws + o);             o += (size_t)192 * KT * 512 * 2;
  __hip_bfloat16*  whhP = (__hip_bfloat16*)(ws + o);             o += (size_t)192 * KT * 512 * 2;
  __hip_bfloat16*  fcP  = (__hip_bfloat16*)(ws + o);             o += (size_t)64  * KT * 512 * 2;
  __hip_bfloat16*  hA0  = (__hip_bfloat16*)(ws + o);             o += (size_t)BD * HD * 2;
  __hip_bfloat16*  hA1  = (__hip_bfloat16*)(ws + o);             o += (size_t)BD * HD * 2;
  float*           hF0  = (float*)(ws + o);                      o += (size_t)BD * HD * 4;
  float*           hF1  = (float*)(ws + o);                      o += (size_t)BD * HD * 4;

  // prep (stream-ordered before the scan)
  init_kernel<<<(BD * SD + 255) / 256, 256, 0, stream>>>(cnt, hA0, hF0, attn);
  pack_x_kernel<<<(SD * KT * 32) / 256, 256, 0, stream>>>(x, xp);
  pack_w_kernel<<<(192 * KT * 32) / 256, 256, 0, stream>>>(w_ih, wihP, 3 * HD, ID);
  pack_w_kernel<<<(192 * KT * 32) / 256, 256, 0, stream>>>(w_hh, whhP, 3 * HD, HD);
  pack_w_kernel<<<(64  * KT * 32) / 256, 256, 0, stream>>>(fc_w, fcP, ID, HD);

  // persistent fused scan
  gru_scan_kernel<<<NWG, TPB, 0, stream>>>(xp, wihP, whhP, fcP, b_ih, b_hh, fc_b,
                                           hA0, hA1, hF0, hF1, out, hidden, cnt);
}
